// RankingLossFunc_48730698941000
// MI455X (gfx1250) — matrix operand order repaired and checked by reading, not compiled, not run on hardware
//
#include <hip/hip_runtime.h>
#include <hip/hip_bf16.h>

#define RN 4096
#define RD 1024
#define RDELTA 0.2f

typedef __bf16 bf16_t;
typedef __attribute__((ext_vector_type(16))) __bf16 v16bf;
typedef __attribute__((ext_vector_type(8)))  __bf16 v8bf;
typedef __attribute__((ext_vector_type(8)))  float  v8f;

// ---------------------------------------------------------------------------
// Kernel 1: per-row L2 norm, normalize, split into bf16 hi/lo pair.
// Blocks 0..RN-1 handle X rows, RN..2RN-1 handle Y rows. 256 thr, 4 elem/thr.
// Also zeroes d_out (block 0, thread 0).
// ---------------------------------------------------------------------------
__global__ __launch_bounds__(256) void normalize_split_kernel(
    const float* __restrict__ X, const float* __restrict__ Y,
    bf16_t* __restrict__ Xh, bf16_t* __restrict__ Xl,
    bf16_t* __restrict__ Yh, bf16_t* __restrict__ Yl,
    float* __restrict__ invX, float* __restrict__ invY,
    float* __restrict__ out)
{
    const int blk  = blockIdx.x;
    const bool isY = blk >= RN;
    const int row  = isY ? (blk - RN) : blk;
    const float* src = (isY ? Y : X) + (size_t)row * RD;
    bf16_t* dh = (isY ? Yh : Xh) + (size_t)row * RD;
    bf16_t* dl = (isY ? Yl : Xl) + (size_t)row * RD;

    const int tid  = threadIdx.x;
    const int lane = tid & 31;
    const int wave = tid >> 5;

    float vals[4];
    float ss = 0.0f;
#pragma unroll
    for (int q = 0; q < 4; ++q) {
        vals[q] = src[tid + q * 256];
        ss += vals[q] * vals[q];
    }
    // wave32 reduce
#pragma unroll
    for (int off = 16; off > 0; off >>= 1)
        ss += __shfl_xor(ss, off, 32);

    __shared__ float sred[8];
    if (lane == 0) sred[wave] = ss;
    __syncthreads();
    float total = 0.0f;
#pragma unroll
    for (int w = 0; w < 8; ++w) total += sred[w];

    const float inv = 1.0f / fmaxf(sqrtf(total), 1e-8f);

#pragma unroll
    for (int q = 0; q < 4; ++q) {
        const float xn = vals[q] * inv;
        const bf16_t h = (bf16_t)xn;
        const float  r = xn - (float)h;
        dh[tid + q * 256] = h;
        dl[tid + q * 256] = (bf16_t)r;
    }

    if (tid == 0) (isY ? invY : invX)[row] = inv;
    if (blk == 0 && tid == 0) out[0] = 0.0f;
}

// ---------------------------------------------------------------------------
// Kernel 2: exact fp32 diagonal: pos[i] = invX[i]*invY[i]*dot(X_i, Y_i)
// ---------------------------------------------------------------------------
__global__ __launch_bounds__(256) void pos_kernel(
    const float* __restrict__ X, const float* __restrict__ Y,
    const float* __restrict__ invX, const float* __restrict__ invY,
    float* __restrict__ pos)
{
    const int row = blockIdx.x;
    const float* xr = X + (size_t)row * RD;
    const float* yr = Y + (size_t)row * RD;
    const int tid  = threadIdx.x;
    const int lane = tid & 31;
    const int wave = tid >> 5;

    float d = 0.0f;
#pragma unroll
    for (int q = 0; q < 4; ++q)
        d += xr[tid + q * 256] * yr[tid + q * 256];
#pragma unroll
    for (int off = 16; off > 0; off >>= 1)
        d += __shfl_xor(d, off, 32);

    __shared__ float sred[8];
    if (lane == 0) sred[wave] = d;
    __syncthreads();
    if (tid == 0) {
        float total = 0.0f;
#pragma unroll
        for (int w = 0; w < 8; ++w) total += sred[w];
        pos[row] = total * invX[row] * invY[row];
    }
}

// ---------------------------------------------------------------------------
// Kernel 3: WMMA GEMM (split-bf16, 3 products per K-chunk) + fused hinge
// reduction. Grid 32x32; block = 256 thr = 8 waves; workgroup tile 128x128 of
// S; each wave computes a 16x128 strip (8 accumulator tiles).
// ---------------------------------------------------------------------------
__global__ __launch_bounds__(256) void rank_wmma_kernel(
    const bf16_t* __restrict__ Xh, const bf16_t* __restrict__ Xl,
    const bf16_t* __restrict__ Yh, const bf16_t* __restrict__ Yl,
    const float* __restrict__ pos, float* __restrict__ out)
{
    const int tid    = threadIdx.x;
    const int lane   = tid & 31;
    const int wave   = tid >> 5;
    const int laneN  = lane & 15;
    const int hiHalf = (lane >= 16) ? 1 : 0;

    const int rowBase = blockIdx.x * 128 + wave * 16;
    const int colBase = blockIdx.y * 128;

    v8f acc[8] = {};

    // A fragment addressing (16x32 bf16 layout):
    //   lanes 0-15 (M = lane):    K = [0..16), contiguous
    //   lanes 16-31 (M = lane-16): K = [16..32), contiguous
    const int aRow = rowBase + laneN;
    const bf16_t* aPtrH = Xh + (size_t)aRow * RD;
    const bf16_t* aPtrL = Xl + (size_t)aRow * RD;
    const int aK0 = hiHalf * 16;

    // B fragment addressing (32x16 bf16 layout): column n = Y row n.
    //   lanes 0-15 hold K=[0..16), lanes 16-31 hold K=[16..32), contiguous.
    const int bK0 = hiHalf * 16;

    for (int k = 0; k < RD; k += 32) {
        v8bf a0h = *(const v8bf*)(aPtrH + k + aK0);
        v8bf a1h = *(const v8bf*)(aPtrH + k + 8 + aK0);
        v8bf a0l = *(const v8bf*)(aPtrL + k + aK0);
        v8bf a1l = *(const v8bf*)(aPtrL + k + 8 + aK0);
        v16bf Ah = __builtin_shufflevector(a0h, a1h,
            0,1,2,3,4,5,6,7,8,9,10,11,12,13,14,15);
        v16bf Al = __builtin_shufflevector(a0l, a1l,
            0,1,2,3,4,5,6,7,8,9,10,11,12,13,14,15);

#pragma unroll
        for (int t = 0; t < 8; ++t) {
            const size_t bRowOff = (size_t)(colBase + t * 16 + laneN) * RD + k + bK0;
            v16bf Bh = *(const v16bf*)(Yh + bRowOff);
            v16bf Bl = *(const v16bf*)(Yl + bRowOff);
            // acc += Ah*Bh + Ah*Bl + Al*Bh   (split-bf16 ~fp32 product)
            acc[t] = __builtin_amdgcn_wmma_f32_16x16x32_bf16(
                false, Ah, false, Bh, (short)0, acc[t], false, false);
            acc[t] = __builtin_amdgcn_wmma_f32_16x16x32_bf16(
                false, Ah, false, Bl, (short)0, acc[t], false, false);
            acc[t] = __builtin_amdgcn_wmma_f32_16x16x32_bf16(
                false, Al, false, Bh, (short)0, acc[t], false, false);
        }
    }

    // Epilogue: hinge + diagonal mask + reduction.
    // C/D layout: VGPR v, lanes 0-15 -> (M=v, N=lane); lanes 16-31 -> (M=v+8).
    float psum = 0.0f;
#pragma unroll
    for (int v = 0; v < 8; ++v) {
        const int i = rowBase + hiHalf * 8 + v;
        const float p = pos[i];
#pragma unroll
        for (int t = 0; t < 8; ++t) {
            const int j = colBase + t * 16 + laneN;
            const float h = fmaxf(0.0f, RDELTA - p + acc[t][v]);
            psum += (i == j) ? 0.0f : h;
        }
    }
#pragma unroll
    for (int off = 16; off > 0; off >>= 1)
        psum += __shfl_xor(psum, off, 32);

    __shared__ float red[8];
    if (lane == 0) red[wave] = psum;
    __syncthreads();
    if (tid == 0) {
        float s = 0.0f;
#pragma unroll
        for (int w = 0; w < 8; ++w) s += red[w];
        atomicAdd(out, s);
    }
}

// ---------------------------------------------------------------------------
extern "C" void kernel_launch(void* const* d_in, const int* in_sizes, int n_in,
                              void* d_out, int out_size, void* d_ws, size_t ws_size,
                              hipStream_t stream) {
    const float* X = (const float*)d_in[0];
    const float* Y = (const float*)d_in[1];
    float* out = (float*)d_out;

    // Workspace layout (needs ~33.6 MB):
    //   Xh, Xl, Yh, Yl : RN*RD bf16 each (8 MiB each)
    //   invX, invY, pos: RN floats each
    const size_t nd = (size_t)RN * RD;
    bf16_t* Xh = (bf16_t*)d_ws;
    bf16_t* Xl = Xh + nd;
    bf16_t* Yh = Xl + nd;
    bf16_t* Yl = Yh + nd;
    float* invX = (float*)(Yl + nd);
    float* invY = invX + RN;
    float* pos  = invY + RN;

    normalize_split_kernel<<<dim3(2 * RN), dim3(256), 0, stream>>>(
        X, Y, Xh, Xl, Yh, Yl, invX, invY, out);
    pos_kernel<<<dim3(RN), dim3(256), 0, stream>>>(X, Y, invX, invY, pos);
    rank_wmma_kernel<<<dim3(32, 32), dim3(256), 0, stream>>>(
        Xh, Xl, Yh, Yl, pos, out);
}
